// ChannelNonlinearSpectralBlock_84370337563129
// MI455X (gfx1250) — matrix operand order repaired and checked by reading, not compiled, not run on hardware
//
#include <hip/hip_runtime.h>
#include <hip/hip_bf16.h>
#include <math.h>
#include <stdint.h>

typedef __attribute__((ext_vector_type(16))) __bf16 v16bf;
typedef __attribute__((ext_vector_type(8)))  __bf16 v8bf;
typedef __attribute__((ext_vector_type(8)))  float  v8f;

#define C_CH 256
#define HW   4096
#define XP   272      // xtile pitch (bf16 elems), multiple of 8 -> 16B aligned chunks
#define BP   40       // B LDS slice pitch (bf16 elems per column), 80 B, 16B aligned chunks
#define BBUF_ELEMS (C_CH * BP)          // 10240 bf16 = 20480 B per buffer
#define WP   18       // wlds pitch (floats), even -> float2 aligned
#define ROWS_PER_BLOCK 64
#define WAVES 4

// shared memory layout (aliased):
//   [0, 34816)           xtile   : 64 x XP bf16   (dead after A-frags/norms)
//   [0, 40960)           bbuf    : 2 x 256 x BP bf16 (B slice double buffer)
//   [40960, 45056)       red     : 4*32*8 floats
//   [45056, 63488)       wlds    : 4*64*WP floats
#define SMEM_BYTES 63488
#define RED_OFF    40960
#define WLDS_OFF   45056

// ---------------------------------------------------------------------------
// Kernel 1: build circulant spectral weights.
//   v = irfft(rfft(u) * exp(i*L*phi))  ==  u (*) w   (circular conv)
//   w[d] = (1/256)[cos(t0) + (-1)^d cos(t128) + 2*sum_{f=1..127} cos(t_f + 2pi f d/256)]
// Store Wt[n][k] = w[(n-k) & 255] as bf16 (B operand of GEMM, column n, depth k).
// ---------------------------------------------------------------------------
__global__ __launch_bounds__(256) void build_wmat_kernel(const float* __restrict__ phi,
                                                         __bf16* __restrict__ Wt) {
  __shared__ float wsh[C_CH];
  const int d = threadIdx.x;
  const float Lc   = 10.0f;
  const float step = 6.28318530717958647692f / 256.0f;
  float s = cosf(Lc * phi[0]) + ((d & 1) ? -1.0f : 1.0f) * cosf(Lc * phi[128]);
  for (int f = 1; f < 128; ++f) {
    s += 2.0f * cosf(fmaf(step, (float)(f * d), Lc * phi[f]));
  }
  wsh[d] = s * (1.0f / 256.0f);
  __syncthreads();
  const int j = d;
  for (int k = 0; k < C_CH; ++k) {
    Wt[j * C_CH + k] = (__bf16)wsh[(j - k) & 255];
  }
}

// ---------------------------------------------------------------------------
// Kernel 2: 64 spatial rows per block, 4 waves, 16 rows per wave.
// ---------------------------------------------------------------------------
__global__ __launch_bounds__(128) void spectral_block_kernel(
    const float* __restrict__ x,
    const float* __restrict__ a0_1, const float* __restrict__ a_1, const float* __restrict__ b_1,
    const float* __restrict__ a0_2, const float* __restrict__ a_2, const float* __restrict__ b_2,
    const float* __restrict__ alpha, const float* __restrict__ beta,
    const __bf16* __restrict__ Wt,
    float* __restrict__ out) {
  __shared__ __align__(16) char smem[SMEM_BYTES];
  __bf16* xtile = (__bf16*)smem;                 // aliases bbuf (xtile dead first)
  __bf16* bbuf  = (__bf16*)smem;
  float*  red   = (float*)(smem + RED_OFF);
  float*  wlds  = (float*)(smem + WLDS_OFF);     // [wave][64][WP]

  const int t   = threadIdx.x;
  const int R0  = blockIdx.x * ROWS_PER_BLOCK;           // global spatial row base
  const int bb  = R0 >> 12;                              // batch index (HW = 4096)
  const int sp0 = R0 & (HW - 1);                         // spatial offset inside image
  const float* xb = x + (size_t)bb * C_CH * HW + sp0;    // element (c,row): xb[c*HW + row]

  // ---- Phase A: coalesced load of 64x256 f32 tile -> bf16 LDS [row][ch] ----
  for (int it = 0; it < 32; ++it) {
    const int c  = it * 8 + (t >> 4);                    // 8 channels / iter
    const int r4 = (t & 15) * 4;                         // 4 consecutive rows / lane
    const float4 v = *(const float4*)(xb + (size_t)c * HW + r4);
    xtile[(r4 + 0) * XP + c] = (__bf16)v.x;
    xtile[(r4 + 1) * XP + c] = (__bf16)v.y;
    xtile[(r4 + 2) * XP + c] = (__bf16)v.z;
    xtile[(r4 + 3) * XP + c] = (__bf16)v.w;
  }
  __syncthreads();

  const int w    = t >> 5;        // wave id (0..3)
  const int l    = t & 31;        // lane id
  const int rloc = l & 15;        // row within wave tile / column within N-tile
  const int h    = l >> 4;        // lane half
  const int rowL = w * 16 + rloc; // row within block tile

  // ---- input row norms: lane half sums 128 channels, combine via shfl_xor ----
  float p0 = 0.0f;
  {
    const int cb0 = h * 128;
#pragma unroll
    for (int cb = 0; cb < 128; cb += 8) {
      v8bf q = *(const v8bf*)&xtile[rowL * XP + cb0 + cb];
#pragma unroll
      for (int i = 0; i < 8; ++i) { float f = (float)q[i]; p0 = fmaf(f, f, p0); }
    }
  }
  const float r0sq = p0 + __shfl_xor(p0, 16, 32);

  // ---- A fragments (ISA 16-bit A layout: lane<16 K{0-7,16-23}, lane>=16 K{8-15,24-31}) ----
  v16bf afr[8];
  const int koff = h * 8;
#pragma unroll
  for (int kc = 0; kc < 8; ++kc) {
    v8bf lo = *(const v8bf*)&xtile[rowL * XP + kc * 32 + koff];
    v8bf hi = *(const v8bf*)&xtile[rowL * XP + kc * 32 + koff + 16];
    afr[kc] = __builtin_shufflevector(lo, hi, 0,1,2,3,4,5,6,7,8,9,10,11,12,13,14,15);
  }
  __syncthreads();   // xtile fully consumed; its LDS space becomes the B double buffer

  // ---- async B-slice staging: chunk kc = Wt[:, kc*32 .. kc*32+32) -> LDS ----
  // Each chunk: 256 cols x 32 bf16 = 16 KB, padded pitch BP. 1024 b128 copies,
  // 128 threads x 8 global_load_async_to_lds_b128 (tracked by ASYNCcnt).
  auto stage = [&](int bsel, int kc) {
#pragma unroll
    for (int i = 0; i < 8; ++i) {
      const int linear = i * 128 + t;          // 0..1023
      const int n = linear >> 2;               // column 0..255
      const int q = linear & 3;                // 16B chunk within 64B row slice
      const __bf16* gp = Wt + (size_t)n * C_CH + kc * 32 + q * 8;
      const unsigned lds = (unsigned)(uintptr_t)(const void*)
          (bbuf + bsel * BBUF_ELEMS + n * BP + q * 8);
      const unsigned long long ga = (unsigned long long)(uintptr_t)gp;
      asm volatile("global_load_async_to_lds_b128 %0, %1, off"
                   :: "v"(lds), "v"(ga) : "memory");
    }
  };

  stage(0, 0);
  stage(1, 1);

  // ---- GEMM: 8 K-chunks x 16 N-tiles of v_wmma_f32_16x16x32_bf16.
  //      B fragments from LDS, software-pipelined 2 N-tiles deep so every
  //      WMMA waits on loads issued >=2 iterations earlier (dscnt<=4). ----
  v8f accs[16];
  const v8f vzero = {0.f,0.f,0.f,0.f,0.f,0.f,0.f,0.f};
#pragma unroll
  for (int nt = 0; nt < 16; ++nt) accs[nt] = vzero;

#pragma unroll
  for (int kc = 0; kc < 8; ++kc) {
    // async loads complete in order: <=8 outstanding => chunk kc has landed
    asm volatile("s_wait_asynccnt 0x8" ::: "memory");
    __syncthreads();                            // copies visible block-wide
    const __bf16* bchunk = bbuf + (kc & 1) * BBUF_ELEMS + (size_t)rloc * BP + h * 8;

    v8bf lo0 = *(const v8bf*)(bchunk);
    v8bf hi0 = *(const v8bf*)(bchunk + 16);
    v8bf lo1 = *(const v8bf*)(bchunk + 16 * BP);
    v8bf hi1 = *(const v8bf*)(bchunk + 16 * BP + 16);
#pragma unroll
    for (int nt = 0; nt < 16; ++nt) {
      v8bf lo2, hi2;
      if (nt + 2 < 16) {
        const __bf16* np = bchunk + (size_t)(nt + 2) * 16 * BP;
        lo2 = *(const v8bf*)np;
        hi2 = *(const v8bf*)(np + 16);
      }
      v16bf bfrag = __builtin_shufflevector(lo0, hi0, 0,1,2,3,4,5,6,7,8,9,10,11,12,13,14,15);
      accs[nt] = __builtin_amdgcn_wmma_f32_16x16x32_bf16(
          false, afr[kc], false, bfrag, (short)0, accs[nt], false, false);
      lo0 = lo1; hi0 = hi1;
      if (nt + 2 < 16) { lo1 = lo2; hi1 = hi2; }
    }
    __syncthreads();                            // all waves done reading this buffer
    stage(kc & 1, (kc + 2) & 7);                // prefetch (tail restages are harmless)
  }

  // ---- output row norms: per-lane sum of squares, reduce across lanes via LDS ----
  float rn[8] = {0.f,0.f,0.f,0.f,0.f,0.f,0.f,0.f};
#pragma unroll
  for (int nt = 0; nt < 16; ++nt)
#pragma unroll
    for (int i = 0; i < 8; ++i) rn[i] = fmaf(accs[nt][i], accs[nt][i], rn[i]);
#pragma unroll
  for (int i = 0; i < 8; ++i) red[(w * 32 + l) * 8 + i] = rn[i];
  __syncthreads();

  float rwsq = 0.0f;
  {
    const int jb = (rloc >= 8) ? 16 : 0;   // rows 0-7 live in lanes 0-15, rows 8-15 in 16-31
    const int ii = rloc & 7;
#pragma unroll
    for (int j = 0; j < 16; ++j) rwsq += red[(w * 32 + jb + j) * 8 + ii];
  }

  // ---- per-row scalar chain (lane == row rloc), all radial gates folded ----
  const float sc = 0.031622776601683794f;            // sqrt(0.001)
  const float r0 = fmaxf(sqrtf(r0sq), 1e-6f);
  const float arg = fminf(sc * r0, 1.0f - 1e-5f);
  const float su  = atanhf(arg) / (sc * r0);         // log-map gate
  const float ru  = fmaxf(su * r0, 1e-6f);           // ||u||
  float f1 = a0_1[0];
#pragma unroll
  for (int n = 1; n <= 16; ++n) {
    float sn, cs; __sincosf(ru * (float)n, &sn, &cs);
    f1 = fmaf(a_1[n - 1], cs, f1);
    f1 = fmaf(b_1[n - 1], sn, f1);
  }
  const float s1  = f1 * su;                         // u1 = s1 * h0
  const float rw  = sqrtf(rwsq);                     // ||W h0||
  const float rv0 = fmaxf(fabsf(s1) * rw, 1e-6f);    // ||v0||
  float f2 = a0_2[0];
#pragma unroll
  for (int n = 1; n <= 16; ++n) {
    float sn, cs; __sincosf(rv0 * (float)n, &sn, &cs);
    f2 = fmaf(a_2[n - 1], cs, f2);
    f2 = fmaf(b_2[n - 1], sn, f2);
  }
  const float s2  = f2 * s1;                         // v1 = s2 * (W h0)
  const float rv1 = fmaxf(fabsf(s2) * rw, 1e-6f);    // ||v1||
  const float g   = tanhf(sc * rv1) / (sc * rv1);    // exp-map gate
  const float stot = alpha[0] * g * s2;
  const float bet  = beta[0];

  // broadcast per-row total scale to the lanes that hold that row in C layout
  float srow[8];
#pragma unroll
  for (int i = 0; i < 8; ++i) srow[i] = __shfl(stot, i + 8 * h, 32);

  // ---- epilogue: scale, transpose via LDS, coalesced store + beta residual ----
  const float* xo = xb + w * 16;
  float* oo = out + (size_t)bb * C_CH * HW + sp0 + w * 16;
  float* wl = wlds + w * 64 * WP;                 // this wave's [64][WP] slice
  for (int gq = 0; gq < 4; ++gq) {
#pragma unroll
    for (int nt2 = 0; nt2 < 4; ++nt2) {
      const int nt = gq * 4 + nt2;
#pragma unroll
      for (int i = 0; i < 8; ++i)
        wl[(nt2 * 16 + rloc) * WP + i + 8 * h] = srow[i] * accs[nt][i];
    }
    __syncthreads();
    for (int it = 0; it < 16; ++it) {
      const int cl = it * 4 + (l >> 3);      // channel within group (0..63)
      const int r2 = (l & 7) * 2;            // row pair
      const float2 v2 = *(const float2*)&wl[cl * WP + r2];
      const size_t adr = (size_t)(gq * 64 + cl) * HW + r2;
      const float2 xv = *(const float2*)(xo + adr);   // L2-hot re-read for residual
      float2 ov;
      ov.x = fmaf(bet, xv.x, v2.x);
      ov.y = fmaf(bet, xv.y, v2.y);
      *(float2*)(oo + adr) = ov;
    }
    __syncthreads();
  }
}

extern "C" void kernel_launch(void* const* d_in, const int* in_sizes, int n_in,
                              void* d_out, int out_size, void* d_ws, size_t ws_size,
                              hipStream_t stream) {
  const float* x     = (const float*)d_in[0];
  const float* a0_1  = (const float*)d_in[1];
  const float* a_1   = (const float*)d_in[2];
  const float* b_1   = (const float*)d_in[3];
  const float* a0_2  = (const float*)d_in[4];
  const float* a_2   = (const float*)d_in[5];
  const float* b_2   = (const float*)d_in[6];
  const float* phi   = (const float*)d_in[7];
  const float* alpha = (const float*)d_in[8];
  const float* beta  = (const float*)d_in[9];

  __bf16* Wt = (__bf16*)d_ws;                       // 256*256*2 = 128 KB scratch

  build_wmat_kernel<<<1, 256, 0, stream>>>(phi, Wt);

  const int M = 32 * 64 * 64;                       // 131072 spatial rows
  spectral_block_kernel<<<M / ROWS_PER_BLOCK, 128, 0, stream>>>(
      x, a0_1, a_1, b_1, a0_2, a_2, b_2, alpha, beta, Wt, (float*)d_out);
}
